// HybridKernelClassifier_65481071395813
// MI455X (gfx1250) — compile-verified, matrix operand order checked by reading, not run on hardware
//
#include <hip/hip_runtime.h>
#include <hip/hip_bf16.h>

typedef __attribute__((ext_vector_type(16))) _Float16 v16h;
typedef __attribute__((ext_vector_type(8)))  float    v8f;
typedef __attribute__((ext_vector_type(2)))  float    v2f;

#ifndef __has_builtin
#define __has_builtin(x) 0
#endif

#define B_   256
#define F_   512
#define E_   64
#define H_   4
#define FFN_ 128

union HF16 { v16h v; float4 f[2]; };

// Load a 16-element f16 fragment: elements 0..7 = 8 contiguous halves at p,
// elements 8..15 = 8 contiguous halves at p+16 (matches ISA 16-bit A/B layout:
// VGPR0-3 hold K=base..base+7, VGPR4-7 hold K=base+16..base+23 for this lane).
__device__ __forceinline__ v16h load_frag16(const _Float16* p) {
    HF16 u;
    u.f[0] = *(const float4*)(p);
    u.f[1] = *(const float4*)(p + 16);
    return u.v;
}

__device__ __forceinline__ float wred_add(float v) {
#pragma unroll
    for (int off = 16; off > 0; off >>= 1) v += __shfl_xor(v, off, 32);
    return v;
}

// ---------------------------------------------------------------- sq[i] = ||x_i||^2
__global__ void __launch_bounds__(256) sq_kernel(const float* __restrict__ x,
                                                 float* __restrict__ sq) {
    int i = threadIdx.x;
    float s = 0.f;
    for (int e = 0; e < F_; ++e) { float v = x[i * F_ + e]; s += v * v; }
    sq[i] = s;
}

// ---------------------------------------------------------------- f32 -> f16 weights
__global__ void __launch_bounds__(256) cvt_kernel(const float* __restrict__ w1,
                                                  const float* __restrict__ w2,
                                                  _Float16* __restrict__ w1h,
                                                  _Float16* __restrict__ w2h) {
    int i = blockIdx.x * 256 + threadIdx.x;       // 16384 threads
    if (i < FFN_ * E_)       w1h[i] = (_Float16)w1[i];
    else                     w2h[i - FFN_ * E_] = (_Float16)w2[i - FFN_ * E_];
}

// ---------------------------------------------------------------- small precompute
__global__ void __launch_bounds__(128)
precompute_kernel(const float* __restrict__ wq, const float* __restrict__ wk,
                  const float* __restrict__ wv, const float* __restrict__ bq,
                  const float* __restrict__ bv, const float* __restrict__ wo,
                  const float* __restrict__ bo,
                  float* __restrict__ ABh, float* __restrict__ bqBh,
                  float* __restrict__ W2, float* __restrict__ b2v) {
    __shared__ float sA[E_], sB[E_], sC[E_];
    int t = threadIdx.x;
    if (t < E_) {
        float sa = 0.f, sb = 0.f, sc = 0.f;
        for (int e = 0; e < E_; ++e) {
            sa += wq[t * E_ + e];
            sb += wk[t * E_ + e];
            sc += wv[t * E_ + e];
        }
        sA[t] = sa; sB[t] = sb; sC[t] = sc;
    }
    __syncthreads();
    if (t < E_) {
#pragma unroll
        for (int h = 0; h < H_; ++h) {
            float s = 0.f;
            for (int d = 0; d < 16; ++d) s += sC[h * 16 + d] * wo[t * E_ + h * 16 + d];
            W2[t * H_ + h] = s;
        }
        float s2 = 0.f;
        for (int e = 0; e < E_; ++e) s2 += bv[e] * wo[t * E_ + e];
        b2v[t] = bo[t] + s2;
    }
    if (t >= 64 && t < 64 + H_) {
        int h = t - 64;
        float ab = 0.f, qb = 0.f;
        for (int d = 0; d < 16; ++d) {
            ab += sA[h * 16 + d] * sB[h * 16 + d];
            qb += bq[h * 16 + d] * sB[h * 16 + d];
        }
        ABh[h] = ab; bqBh[h] = qb;
    }
}

// ---------------------------------------------------------------- K = exp(-d2), Gram via WMMA
__global__ void __launch_bounds__(256)
gram_kernel(const float* __restrict__ x, const float* __restrict__ sq,
            float* __restrict__ Kmat) {
    int gw    = (blockIdx.x * 256 + threadIdx.x) >> 5;  // 0..255 -> one 16x16 tile per wave
    int lane  = threadIdx.x & 31;
    int l15   = lane & 15;
    int khalf = lane >> 4;
    int tm = (gw >> 4) * 16;
    int tn = (gw & 15) * 16;
    int row = tm + l15;   // A-matrix row  (M)
    int col = tn + l15;   // B-matrix col  (N)
    v8f acc = {};
#if __has_builtin(__builtin_amdgcn_wmma_f32_16x16x4_f32)
    for (int k = 0; k < F_; k += 4) {
        int kb = k + khalf * 2;                       // f32 A 16x4: v0=K0/K2, v1=K1/K3
        v2f a  = *(const v2f*)(x + row * F_ + kb);
        v2f bm = *(const v2f*)(x + col * F_ + kb);
        acc = __builtin_amdgcn_wmma_f32_16x16x4_f32(false, a, false, bm,
                                                    (short)0, acc, false, false);
    }
#else
    for (int k = 0; k < F_; k += 32) {
        v16h a, bm;
#pragma unroll
        for (int i = 0; i < 16; ++i) {
            int kk = k + khalf * 8 + i + ((i >= 8) ? 8 : 0);
            a[i]  = (_Float16)x[row * F_ + kk];
            bm[i] = (_Float16)x[col * F_ + kk];
        }
        acc = __builtin_amdgcn_wmma_f32_16x16x32_f16(false, a, false, bm,
                                                     (short)0, acc, false, false);
    }
#endif
    float sqn = sq[col];
#pragma unroll
    for (int j = 0; j < 8; ++j) {                     // C layout: lane holds N=col, M=khalf*8+j
        int m = tm + khalf * 8 + j;
        float d2 = fmaxf(sq[m] + sqn - 2.0f * acc[j], 0.0f);
        Kmat[m * B_ + col] = __expf(-d2);             // GAMMA = 1.0
    }
}

// ---------------------------------------------------------------- m[b,h,q] softmax moments
__global__ void __launch_bounds__(256)
attn_m_kernel(const float* __restrict__ Kmat, const float* __restrict__ ABh,
              const float* __restrict__ bqBh, float* __restrict__ mOut) {
    __shared__ float kr[B_];
    __shared__ float smx[B_];
    __shared__ float smn[B_];
    int b = blockIdx.x, t = threadIdx.x;
    float kv = Kmat[b * B_ + t];
    kr[t] = kv; smx[t] = kv; smn[t] = kv;
    __syncthreads();
    for (int s = 128; s > 0; s >>= 1) {
        if (t < s) {
            smx[t] = fmaxf(smx[t], smx[t + s]);
            smn[t] = fminf(smn[t], smn[t + s]);
        }
        __syncthreads();
    }
    float kmax = smx[0], kmin = smn[0];
#pragma unroll
    for (int h = 0; h < H_; ++h) {
        float c1 = 0.25f * (kv * ABh[h] + bqBh[h]);   // 1/sqrt(DH) = 0.25
        float mx = fmaxf(c1 * kmax, c1 * kmin);
        float s0 = 0.f, s1 = 0.f;
        for (int k = 0; k < B_; ++k) {
            float kk = kr[k];
            float e = __expf(c1 * kk - mx);
            s0 += e; s1 += e * kk;
        }
        mOut[(b * H_ + h) * B_ + t] = s1 / s0;
    }
}

// ---------------------------------------------------------------- fused LN1+FFN+LN2+pool
__global__ void __launch_bounds__(128)
ffn_kernel(const float* __restrict__ Kmat, const float* __restrict__ mOut,
           const float* __restrict__ W2, const float* __restrict__ b2v,
           const float* __restrict__ ln1g, const float* __restrict__ ln1b,
           const _Float16* __restrict__ w1h, const float* __restrict__ fb1,
           const _Float16* __restrict__ w2h, const float* __restrict__ fb2,
           const float* __restrict__ ln2g, const float* __restrict__ ln2b,
           float* __restrict__ poolPart) {
    __shared__ __align__(16) _Float16 sH1[4][16][E_];
    __shared__ __align__(16) _Float16 sT[4][16][FFN_];
    __shared__ float sH2[4][16][E_];
    __shared__ float sPool[E_];

    const int b     = blockIdx.x >> 2;
    const int chunk = blockIdx.x & 3;
    const int wv    = threadIdx.x >> 5;
    const int lane  = threadIdx.x & 31;
    const int l15   = lane & 15;
    const int khalf = lane >> 4;
    const int q0    = chunk * 64 + wv * 16;

    if (threadIdx.x < E_) sPool[threadIdx.x] = 0.0f;

    const int e0 = 2 * lane, e1 = 2 * lane + 1;   // lane owns 2 feature columns
    float W2a[H_], W2b[H_];
#pragma unroll
    for (int h = 0; h < H_; ++h) { W2a[h] = W2[e0 * H_ + h]; W2b[h] = W2[e1 * H_ + h]; }
    const float b2a = b2v[e0], b2b = b2v[e1];
    const float g1a = ln1g[e0], g1b = ln1g[e1], o1a = ln1b[e0], o1b = ln1b[e1];
    const float g2a = ln2g[e0], g2b = ln2g[e1], o2a = ln2b[e0], o2b = ln2b[e1];

    // Stage A: base = K[b,q] + attn_out ; LN1 ; store f16 tile
    for (int r = 0; r < 16; ++r) {
        int q = q0 + r;
        float Kbq = Kmat[b * B_ + q];
        float m0 = mOut[(b * H_ + 0) * B_ + q];
        float m1 = mOut[(b * H_ + 1) * B_ + q];
        float m2 = mOut[(b * H_ + 2) * B_ + q];
        float m3 = mOut[(b * H_ + 3) * B_ + q];
        float va = Kbq + b2a + m0 * W2a[0] + m1 * W2a[1] + m2 * W2a[2] + m3 * W2a[3];
        float vb = Kbq + b2b + m0 * W2b[0] + m1 * W2b[1] + m2 * W2b[2] + m3 * W2b[3];
        float mu  = wred_add(va + vb) * (1.0f / 64.0f);
        float da = va - mu, db = vb - mu;
        float var = wred_add(da * da + db * db) * (1.0f / 64.0f);
        float inv = rsqrtf(var + 1e-5f);
        sH1[wv][r][e0] = (_Float16)(da * inv * g1a + o1a);
        sH1[wv][r][e1] = (_Float16)(db * inv * g1b + o1b);
    }
    __syncthreads();

    // Stage B: t = relu(h1 @ w1^T + b1)  [16x64]x[64x128]
#pragma unroll
    for (int nt = 0; nt < 8; ++nt) {
        v8f acc = {};
#pragma unroll
        for (int kk = 0; kk < E_; kk += 32) {
            v16h a  = load_frag16(&sH1[wv][l15][kk + khalf * 8]);
            v16h bm = load_frag16(w1h + (nt * 16 + l15) * E_ + kk + khalf * 8);
            acc = __builtin_amdgcn_wmma_f32_16x16x32_f16(false, a, false, bm,
                                                         (short)0, acc, false, false);
        }
        int n = nt * 16 + l15;
        float bias = fb1[n];
#pragma unroll
        for (int j = 0; j < 8; ++j) {
            int mr = khalf * 8 + j;
            sT[wv][mr][n] = (_Float16)fmaxf(acc[j] + bias, 0.0f);
        }
    }
    __syncthreads();

    // Stage C: f = t @ w2^T + b2 ; residual h1 + f   [16x128]x[128x64]
#pragma unroll
    for (int nt = 0; nt < 4; ++nt) {
        v8f acc = {};
#pragma unroll
        for (int kk = 0; kk < FFN_; kk += 32) {
            v16h a  = load_frag16(&sT[wv][l15][kk + khalf * 8]);
            v16h bm = load_frag16(w2h + (nt * 16 + l15) * FFN_ + kk + khalf * 8);
            acc = __builtin_amdgcn_wmma_f32_16x16x32_f16(false, a, false, bm,
                                                         (short)0, acc, false, false);
        }
        int o = nt * 16 + l15;
        float bias = fb2[o];
#pragma unroll
        for (int j = 0; j < 8; ++j) {
            int mr = khalf * 8 + j;
            sH2[wv][mr][o] = (float)sH1[wv][mr][o] + acc[j] + bias;
        }
    }
    __syncthreads();

    // Stage D: LN2 + pooling accumulation
    float pa = 0.f, pb = 0.f;
    for (int r = 0; r < 16; ++r) {
        float va = sH2[wv][r][e0], vb = sH2[wv][r][e1];
        float mu  = wred_add(va + vb) * (1.0f / 64.0f);
        float da = va - mu, db = vb - mu;
        float var = wred_add(da * da + db * db) * (1.0f / 64.0f);
        float inv = rsqrtf(var + 1e-5f);
        pa += da * inv * g2a + o2a;
        pb += db * inv * g2b + o2b;
    }
    atomicAdd(&sPool[e0], pa);
    atomicAdd(&sPool[e1], pb);
    __syncthreads();
    if (threadIdx.x < E_)
        poolPart[blockIdx.x * E_ + threadIdx.x] = sPool[threadIdx.x];
}

// ---------------------------------------------------------------- classifier
__global__ void __launch_bounds__(256)
clf_kernel(const float* __restrict__ poolPart, const float* __restrict__ w1,
           const float* __restrict__ b1, const float* __restrict__ w2,
           const float* __restrict__ b2, float* __restrict__ out) {
    int b = threadIdx.x;
    float pe[E_];
#pragma unroll 4
    for (int e = 0; e < E_; ++e) {
        float s = poolPart[(b * 4 + 0) * E_ + e] + poolPart[(b * 4 + 1) * E_ + e] +
                  poolPart[(b * 4 + 2) * E_ + e] + poolPart[(b * 4 + 3) * E_ + e];
        pe[e] = s * (1.0f / 256.0f);
    }
    float o0 = b2[0], o1 = b2[1];
    for (int o = 0; o < 32; ++o) {
        float s = b1[o];
        for (int e = 0; e < E_; ++e) s += pe[e] * w1[o * E_ + e];
        s = fmaxf(s, 0.0f);
        o0 += s * w2[0 * 32 + o];
        o1 += s * w2[1 * 32 + o];
    }
    out[b * 2 + 0] = o0;
    out[b * 2 + 1] = o1;
}

// ---------------------------------------------------------------- launch
extern "C" void kernel_launch(void* const* d_in, const int* in_sizes, int n_in,
                              void* d_out, int out_size, void* d_ws, size_t ws_size,
                              hipStream_t stream) {
    (void)in_sizes; (void)n_in; (void)out_size; (void)ws_size;
    const float* x    = (const float*)d_in[0];
    const float* wq   = (const float*)d_in[1];
    const float* bq   = (const float*)d_in[2];
    const float* wk   = (const float*)d_in[3];
    const float* wv   = (const float*)d_in[5];
    const float* bv   = (const float*)d_in[6];
    const float* wo   = (const float*)d_in[7];
    const float* bo   = (const float*)d_in[8];
    const float* ln1g = (const float*)d_in[9];
    const float* ln1b = (const float*)d_in[10];
    const float* fw1  = (const float*)d_in[11];
    const float* fb1  = (const float*)d_in[12];
    const float* fw2  = (const float*)d_in[13];
    const float* fb2  = (const float*)d_in[14];
    const float* ln2g = (const float*)d_in[15];
    const float* ln2b = (const float*)d_in[16];
    const float* cw1  = (const float*)d_in[17];
    const float* cb1  = (const float*)d_in[18];
    const float* cw2  = (const float*)d_in[19];
    const float* cb2  = (const float*)d_in[20];
    float* out = (float*)d_out;

    char* ws = (char*)d_ws;
    float*    Kmat  = (float*)(ws + 0);            // 256*256*4      = 262144
    float*    sq    = (float*)(ws + 262144);       // 256*4          = 1024
    float*    mOut  = (float*)(ws + 263168);       // 4*256*256*4    = 1048576
    float*    ABh   = (float*)(ws + 1311744);      // 4
    float*    bqBh  = ABh + 4;                     // 4
    float*    W2    = bqBh + 4;                    // 64*4 = 256
    float*    b2v   = W2 + 256;                    // 64
    _Float16* w1h   = (_Float16*)(ws + 1313824);   // 8192 halves
    _Float16* w2h   = w1h + FFN_ * E_;             // 8192 halves
    float*    poolP = (float*)(ws + 1346592);      // 1024*64*4 = 262144

    sq_kernel<<<1, 256, 0, stream>>>(x, sq);
    cvt_kernel<<<64, 256, 0, stream>>>(fw1, fw2, w1h, w2h);
    precompute_kernel<<<1, 128, 0, stream>>>(wq, wk, wv, bq, bv, wo, bo,
                                             ABh, bqBh, W2, b2v);
    gram_kernel<<<32, 256, 0, stream>>>(x, sq, Kmat);
    attn_m_kernel<<<256, 256, 0, stream>>>(Kmat, ABh, bqBh, mOut);
    ffn_kernel<<<1024, 128, 0, stream>>>(Kmat, mOut, W2, b2v, ln1g, ln1b,
                                         w1h, fb1, w2h, fb2, ln2g, ln2b, poolP);
    clf_kernel<<<1, 256, 0, stream>>>(poolP, cw1, cb1, cw2, cb2, out);
}
